// AttnBlock_63178968924624
// MI455X (gfx1250) — compile-verified, matrix operand order checked by reading, not run on hardware
//
#include <hip/hip_runtime.h>

typedef __attribute__((ext_vector_type(16))) __bf16 v16bf;
typedef __attribute__((ext_vector_type(8)))  float  v8f;

#define WMMA_BF16(a, b, c) \
    __builtin_amdgcn_wmma_f32_16x16x32_bf16(false, (a), false, (b), (short)0, (c), false, false)

// Constants for this problem
#define BATCH 8
#define SEQ   1024
#define CH    512
#define NHEAD 8
#define HDIM  64
#define NQKV  1536              // NHEAD * 3 * HDIM
#define QK_SCALE 0.35355339059327373f   // 64^(-1/4)

struct alignas(16) B16x8 { unsigned int w[4]; };   // 8 bf16 (POD, union-safe)
union U4 { B16x8 u; __bf16 h[8]; unsigned int w[4]; };
union FragU { v16bf v; unsigned int u[8]; __bf16 h[16]; };

// CDNA5 async copy: global -> LDS, 16 bytes per lane, tracked by ASYNCcnt.
// GV addressing: 64-bit global address in a VGPR pair, LDS byte address in a
// VGPR (low 32 bits of the generic pointer to a __shared__ object are the LDS
// address per the flat-aperture rules).
__device__ __forceinline__ void async_ld_b128(void* lds, const void* gptr) {
    unsigned lds_addr = (unsigned)(size_t)lds;
    unsigned long long gaddr = (unsigned long long)(size_t)gptr;
    asm volatile("global_load_async_to_lds_b128 %0, %1, off"
                 :: "v"(lds_addr), "v"(gaddr) : "memory");
}
__device__ __forceinline__ void wait_async0() {
    asm volatile("s_wait_asynccnt 0x0" ::: "memory");
}

// Gather one 16x32 bf16 WMMA operand fragment (A: row = lane&15; B: col =
// lane&15 with [col][k] storage). Lanes 0-15 kbase=0, 16-31 kbase=8;
// element e -> k = kbase + (e&7) + (e>>3)*16, fetched as 8 dword pairs
// (compiler coalesces to ds_load_b128).
__device__ __forceinline__ v16bf frag_ld(const __bf16* rowp) {
    const int hl = (threadIdx.x >> 4) & 1;
    FragU r;
    const unsigned int* q = (const unsigned int*)(rowp + hl * 8);
#pragma unroll
    for (int t = 0; t < 8; ++t) r.u[t] = q[(t & 3) + ((t >> 2) << 3)];
    return r.v;
}

__device__ __forceinline__ v8f vzero8() {
    v8f z = {0.f, 0.f, 0.f, 0.f, 0.f, 0.f, 0.f, 0.f};
    return z;
}

// ---------------------------------------------------------------------------
// Kernel 1: GroupNorm (flax semantics: mean/var over (S, C/G) per batch/group)
// grid = BATCH*32 blocks, 256 threads
// ---------------------------------------------------------------------------
__global__ void __launch_bounds__(256) gn_kernel(const float* __restrict__ x,
                                                 const float* __restrict__ gamma,
                                                 const float* __restrict__ beta,
                                                 __bf16* __restrict__ xn) {
    const int b = blockIdx.x >> 5;
    const int g = blockIdx.x & 31;
    const float* xb = x + (size_t)b * SEQ * CH + g * 16;
    __bf16* xnb = xn + (size_t)b * SEQ * CH + g * 16;

    float s0 = 0.f, s1 = 0.f;
    for (int k = threadIdx.x; k < SEQ * 16; k += 256) {
        int s = k >> 4, j = k & 15;
        float v = xb[s * CH + j];
        s0 += v; s1 += v * v;
    }
    __shared__ float r0[256], r1[256];
    r0[threadIdx.x] = s0; r1[threadIdx.x] = s1;
    __syncthreads();
    for (int off = 128; off > 0; off >>= 1) {
        if ((int)threadIdx.x < off) {
            r0[threadIdx.x] += r0[threadIdx.x + off];
            r1[threadIdx.x] += r1[threadIdx.x + off];
        }
        __syncthreads();
    }
    const float mean = r0[0] * (1.0f / (SEQ * 16));
    const float var  = r1[0] * (1.0f / (SEQ * 16)) - mean * mean;
    const float rstd = rsqrtf(var + 1e-5f);

    for (int k = threadIdx.x; k < SEQ * 16; k += 256) {
        int s = k >> 4, j = k & 15;
        float v = (xb[s * CH + j] - mean) * rstd * gamma[g * 16 + j] + beta[g * 16 + j];
        xnb[s * CH + j] = (__bf16)v;
    }
}

// ---------------------------------------------------------------------------
// Kernel 2: QKV projection GEMM  (M=8192, N=1536, K=512), bf16 WMMA, f32 acc
// Block tile 128x64, 8 waves (4x2), 32x32 per wave -> 4 WMMAs per K-step.
// grid = (64, 24), 256 threads
// ---------------------------------------------------------------------------
__global__ void __launch_bounds__(256) qkv_gemm(const __bf16* __restrict__ xn,
                                                const float* __restrict__ Wqkv,
                                                const float* __restrict__ bias,
                                                __bf16* __restrict__ Q,
                                                __bf16* __restrict__ K,
                                                __bf16* __restrict__ V) {
    const int m0 = blockIdx.x * 128;
    const int n0 = blockIdx.y * 64;
    const int t = threadIdx.x;
    const int w = t >> 5, lane = t & 31, l16 = lane & 15, hl = (lane >> 4) & 1;
    const int mw = (w >> 1) * 32, nw = (w & 1) * 32;

    __shared__ __bf16 At[128 * 32];  // [m][k]
    __shared__ __bf16 Bt[64 * 32];   // [n][k] (transposed for pair-contiguous k)

    v8f acc[2][2] = {{vzero8(), vzero8()}, {vzero8(), vzero8()}};

    for (int ks = 0; ks < CH; ks += 32) {
        // A tile: 128x32 bf16, async global->LDS, 2 x 16B per thread
#pragma unroll
        for (int i = 0; i < 2; ++i) {
            int c = t + i * 256;
            int row = c >> 2, c8 = (c & 3) * 8;
            async_ld_b128(&At[row * 32 + c8],
                          xn + (size_t)(m0 + row) * CH + ks + c8);
        }
        // B tile: load f32 weights, convert to bf16, store transposed [n][k]
        {
            int kk = t >> 3, n8 = (t & 7) * 8;
            const float* wp = Wqkv + (size_t)(ks + kk) * NQKV + n0 + n8;
#pragma unroll
            for (int j = 0; j < 8; ++j) Bt[(n8 + j) * 32 + kk] = (__bf16)wp[j];
        }
        wait_async0();
        __syncthreads();

        v16bf af0 = frag_ld(&At[(mw + l16) * 32]);
        v16bf af1 = frag_ld(&At[(mw + 16 + l16) * 32]);
        v16bf bf0 = frag_ld(&Bt[(nw + l16) * 32]);
        v16bf bf1 = frag_ld(&Bt[(nw + 16 + l16) * 32]);
        acc[0][0] = WMMA_BF16(af0, bf0, acc[0][0]);
        acc[0][1] = WMMA_BF16(af0, bf1, acc[0][1]);
        acc[1][0] = WMMA_BF16(af1, bf0, acc[1][0]);
        acc[1][1] = WMMA_BF16(af1, bf1, acc[1][1]);
        __syncthreads();
    }

    // Epilogue: + bias, scale q/k, scatter into (B, NH, S, HD) bf16
#pragma unroll
    for (int cf = 0; cf < 2; ++cf) {
        const int n = n0 + nw + cf * 16 + l16;
        const int head = n / 192;
        const int j = n - head * 192;
        const float bv = bias[n];
#pragma unroll
        for (int mi = 0; mi < 2; ++mi) {
#pragma unroll
            for (int r = 0; r < 8; ++r) {
                const int m = m0 + mw + mi * 16 + r + hl * 8;
                const int bb = m >> 10, s = m & 1023;
                float val = acc[mi][cf][r] + bv;
                const size_t base = ((size_t)(bb * NHEAD + head) * SEQ + s) * HDIM;
                if (j < 64)       Q[base + j]         = (__bf16)(val * QK_SCALE);
                else if (j < 128) K[base + (j - 64)]  = (__bf16)(val * QK_SCALE);
                else              V[base + (j - 128)] = (__bf16)val;
            }
        }
    }
}

// ---------------------------------------------------------------------------
// Kernel 3: flash attention per (b, h), 128 queries per block, 32-key tiles
// grid = B*NH*(S/128) = 512 blocks, 256 threads (8 waves x 16 query rows)
// ---------------------------------------------------------------------------
__global__ void __launch_bounds__(256) attn_kernel(const __bf16* __restrict__ Q,
                                                   const __bf16* __restrict__ K,
                                                   const __bf16* __restrict__ V,
                                                   __bf16* __restrict__ O) {
    const int bid = blockIdx.x;
    const int qt = bid & 7, h = (bid >> 3) & 7, b = bid >> 6;
    const int t = threadIdx.x, w = t >> 5, lane = t & 31;
    const int l16 = lane & 15, hl = (lane >> 4) & 1;
    const int q0 = qt * 128 + w * 16;

    __shared__ __bf16 Kt[32 * 64];        // [key][d]
    __shared__ __bf16 Vt[64 * 32];        // [d][key] (transposed)
    __shared__ __bf16 Pt[8 * 16 * 32];    // per-wave P patch [row][key]

    const size_t hoff = (size_t)(b * NHEAD + h) * SEQ * HDIM;
    const __bf16* Qg = Q + hoff;
    const __bf16* Kg = K + hoff;
    const __bf16* Vg = V + hoff;
    __bf16* Pw = &Pt[w * 512];

    // Q fragments held in registers for the whole pass (16 rows x 64 d)
    v16bf aq0 = frag_ld(Qg + (size_t)(q0 + l16) * HDIM + 0);
    v16bf aq1 = frag_ld(Qg + (size_t)(q0 + l16) * HDIM + 32);

    v8f o[4] = {vzero8(), vzero8(), vzero8(), vzero8()};
    float m[8], l[8];
#pragma unroll
    for (int r = 0; r < 8; ++r) { m[r] = -1e30f; l[r] = 0.f; }

    for (int kt = 0; kt < SEQ; kt += 32) {
        // K tile: verbatim async global->LDS; V tile: manual transposed store
        {
            int key = t >> 3, d8 = (t & 7) * 8;
            async_ld_b128(&Kt[key * 64 + d8],
                          Kg + (size_t)(kt + key) * HDIM + d8);
            U4 vv; vv.u = *(const B16x8*)(Vg + (size_t)(kt + key) * HDIM + d8);
#pragma unroll
            for (int j = 0; j < 8; ++j) Vt[(d8 + j) * 32 + key] = vv.h[j];
        }
        wait_async0();
        __syncthreads();

        // scores: 16q x 32k, two 16x16 fragments, K-dim = 64 (2 WMMAs each)
        v8f sf[2];
#pragma unroll
        for (int cf = 0; cf < 2; ++cf) {
            v16bf bk0 = frag_ld(&Kt[(cf * 16 + l16) * 64 + 0]);
            v16bf bk1 = frag_ld(&Kt[(cf * 16 + l16) * 64 + 32]);
            v8f s = vzero8();
            s = WMMA_BF16(aq0, bk0, s);
            s = WMMA_BF16(aq1, bk1, s);
            sf[cf] = s;
        }

        // online softmax (row reductions across the 16 lanes of each half)
#pragma unroll
        for (int r = 0; r < 8; ++r) {
            float c = fmaxf(sf[0][r], sf[1][r]);
#pragma unroll
            for (int off = 1; off < 16; off <<= 1) c = fmaxf(c, __shfl_xor(c, off, 32));
            const float mn = fmaxf(m[r], c);
            const float sc = __expf(m[r] - mn);
            m[r] = mn;
            const float p0 = __expf(sf[0][r] - mn);
            const float p1 = __expf(sf[1][r] - mn);
            l[r] = l[r] * sc + p0 + p1;
            const int row = r + hl * 8;
            Pw[row * 32 + l16]      = (__bf16)p0;
            Pw[row * 32 + 16 + l16] = (__bf16)p1;
#pragma unroll
            for (int f = 0; f < 4; ++f) o[f][r] = o[f][r] * sc;
        }

        // wave-local LDS round trip: D-layout P -> A-layout fragment
        asm volatile("s_wait_dscnt 0x0" ::: "memory");
        v16bf ap = frag_ld(&Pw[l16 * 32]);
#pragma unroll
        for (int f = 0; f < 4; ++f) {
            v16bf bv = frag_ld(&Vt[(f * 16 + l16) * 32]);
            o[f] = WMMA_BF16(ap, bv, o[f]);
        }
        __syncthreads();
    }

    // finalize: divide by row sums, write O as (B, S, NH, HD) bf16
#pragma unroll
    for (int r = 0; r < 8; ++r) {
        float ls = l[r];
#pragma unroll
        for (int off = 1; off < 16; off <<= 1) ls += __shfl_xor(ls, off, 32);
        const float inv = 1.0f / ls;
        const int qrow = q0 + r + hl * 8;
        const size_t obase = (((size_t)b * SEQ + qrow) * NHEAD + h) * HDIM;
#pragma unroll
        for (int f = 0; f < 4; ++f)
            O[obase + f * 16 + l16] = (__bf16)(o[f][r] * inv);
    }
}

// ---------------------------------------------------------------------------
// Kernel 4: output projection GEMM (M=8192, N=512, K=512) + bias + residual
// Block tile 128x64, 32x32 per wave. grid = (64, 8), 256 threads
// ---------------------------------------------------------------------------
__global__ void __launch_bounds__(256) out_gemm(const __bf16* __restrict__ Oattn,
                                                const float* __restrict__ Wo,
                                                const float* __restrict__ bias,
                                                const float* __restrict__ x,
                                                float* __restrict__ out) {
    const int m0 = blockIdx.x * 128;
    const int n0 = blockIdx.y * 64;
    const int t = threadIdx.x;
    const int w = t >> 5, lane = t & 31, l16 = lane & 15, hl = (lane >> 4) & 1;
    const int mw = (w >> 1) * 32, nw = (w & 1) * 32;

    __shared__ __bf16 At[128 * 32];
    __shared__ __bf16 Bt[64 * 32];

    v8f acc[2][2] = {{vzero8(), vzero8()}, {vzero8(), vzero8()}};

    for (int ks = 0; ks < CH; ks += 32) {
#pragma unroll
        for (int i = 0; i < 2; ++i) {
            int c = t + i * 256;
            int row = c >> 2, c8 = (c & 3) * 8;
            async_ld_b128(&At[row * 32 + c8],
                          Oattn + (size_t)(m0 + row) * CH + ks + c8);
        }
        {
            int kk = t >> 3, n8 = (t & 7) * 8;
            const float* wp = Wo + (size_t)(ks + kk) * CH + n0 + n8;
#pragma unroll
            for (int j = 0; j < 8; ++j) Bt[(n8 + j) * 32 + kk] = (__bf16)wp[j];
        }
        wait_async0();
        __syncthreads();

        v16bf af0 = frag_ld(&At[(mw + l16) * 32]);
        v16bf af1 = frag_ld(&At[(mw + 16 + l16) * 32]);
        v16bf bf0 = frag_ld(&Bt[(nw + l16) * 32]);
        v16bf bf1 = frag_ld(&Bt[(nw + 16 + l16) * 32]);
        acc[0][0] = WMMA_BF16(af0, bf0, acc[0][0]);
        acc[0][1] = WMMA_BF16(af0, bf1, acc[0][1]);
        acc[1][0] = WMMA_BF16(af1, bf0, acc[1][0]);
        acc[1][1] = WMMA_BF16(af1, bf1, acc[1][1]);
        __syncthreads();
    }

#pragma unroll
    for (int cf = 0; cf < 2; ++cf) {
        const int n = n0 + nw + cf * 16 + l16;
        const float bv = bias[n];
#pragma unroll
        for (int mi = 0; mi < 2; ++mi) {
#pragma unroll
            for (int r = 0; r < 8; ++r) {
                const int m = m0 + mw + mi * 16 + r + hl * 8;
                const size_t idx = (size_t)m * CH + n;
                out[idx] = acc[mi][cf][r] + bv + x[idx];   // SKIP_SCALE == 1.0
            }
        }
    }
}

// ---------------------------------------------------------------------------
extern "C" void kernel_launch(void* const* d_in, const int* in_sizes, int n_in,
                              void* d_out, int out_size, void* d_ws, size_t ws_size,
                              hipStream_t stream) {
    const float* x     = (const float*)d_in[0];
    const float* gamma = (const float*)d_in[1];
    const float* beta  = (const float*)d_in[2];
    const float* qkvW  = (const float*)d_in[3];
    const float* qkvB  = (const float*)d_in[4];
    const float* outW  = (const float*)d_in[5];
    const float* outB  = (const float*)d_in[6];
    float* out = (float*)d_out;

    char* ws = (char*)d_ws;
    const size_t TEN = (size_t)BATCH * SEQ * CH * sizeof(__bf16);  // 8 MB
    __bf16* xn = (__bf16*)(ws + 0 * TEN);
    __bf16* Q  = (__bf16*)(ws + 1 * TEN);
    __bf16* K  = (__bf16*)(ws + 2 * TEN);
    __bf16* V  = (__bf16*)(ws + 3 * TEN);
    __bf16* O  = (__bf16*)(ws + 4 * TEN);

    gn_kernel<<<BATCH * 32, 256, 0, stream>>>(x, gamma, beta, xn);
    qkv_gemm<<<dim3(BATCH * SEQ / 128, NQKV / 64), 256, 0, stream>>>(xn, qkvW, qkvB, Q, K, V);
    attn_kernel<<<BATCH * NHEAD * (SEQ / 128), 256, 0, stream>>>(Q, K, V, O);
    out_gemm<<<dim3(BATCH * SEQ / 128, CH / 64), 256, 0, stream>>>(O, outW, outB, x, out);
}
